// CausalSelfAttention_66924180407283
// MI455X (gfx1250) — matrix operand order repaired and checked by reading, not compiled, not run on hardware
//
#include <hip/hip_runtime.h>
#include <hip/hip_bf16.h>

// ---------------------------------------------------------------------------
// CausalSelfAttention on MI455X (gfx1250), wave32, WMMA bf16 path.
//   B=4, T=2048, C=1024, H=16, D=64
// Pipeline:
//   k1: qkv = x @ w_qkv + b_qkv  -> q/k/v bf16 in [B,H,T,D] workspace
//   k2: flash attention (online softmax) -> att bf16 in [B,T,C] workspace
//        (K tile staged by the Tensor Data Mover: tensor_load_to_lds)
//   k3: out = att @ w_out + b_out -> f32 d_out (A tile staged by TDM, 2D)
// All matmuls use v_wmma_f32_16x16x32_bf16 (16x16x32, f32 accum).
// ---------------------------------------------------------------------------

typedef __attribute__((ext_vector_type(16))) __bf16 v16bf;
typedef __attribute__((ext_vector_type(8)))  float  v8f;
typedef __attribute__((ext_vector_type(4)))  unsigned int u32x4;
typedef __attribute__((ext_vector_type(8)))  int         i32x8;
typedef __attribute__((ext_vector_type(4)))  int         i32x4;

union FragBF {
    v16bf v;
    uint4 q[2];
};

__device__ inline unsigned int pack2_bf16(float a, float b) {
#if __has_builtin(__builtin_amdgcn_cvt_pk_bf16_f32)
    typedef __attribute__((ext_vector_type(2))) __bf16 v2bf;
    v2bf p = __builtin_amdgcn_cvt_pk_bf16_f32(a, b);
    return __builtin_bit_cast(unsigned int, p);
#else
    unsigned int ua = __builtin_bit_cast(unsigned int, a);
    unsigned int ub = __builtin_bit_cast(unsigned int, b);
    unsigned int ra = (ua + 0x7FFFu + ((ua >> 16) & 1u)) >> 16;
    unsigned int rb = (ub + 0x7FFFu + ((ub >> 16) & 1u)) >> 16;
    return (ra & 0xFFFFu) | (rb << 16);
#endif
}

__device__ inline unsigned short f32_to_bf16(float f) {
    return (unsigned short)(pack2_bf16(f, f) & 0xFFFFu);
}

__device__ inline v8f wmma_bf16(const FragBF& a, const FragBF& b, v8f c) {
    // (neg_a, A, neg_b, B, c_mod, C, reuse_a, reuse_b)
    return __builtin_amdgcn_wmma_f32_16x16x32_bf16(false, a.v, false, b.v,
                                                   (short)0, c, false, false);
}

__device__ inline v8f v8f_zero() {
    v8f z = {0.f, 0.f, 0.f, 0.f, 0.f, 0.f, 0.f, 0.f};
    return z;
}

// ---------------------------------------------------------------------------
// Tensor Data Mover: 2D tile load (bf16 elements) from global to LDS.
// Builds the D# per CDNA5 ISA 8.3/8.4: group0 {flags, lds_addr, global_addr,
// type=2}, group1 {data_size=2B, tensor dims, tile dims, dim0 stride}.
// Groups 2..4 zero (tensor <= 2D, no gather/iterate).
// Caller must be a single wave; completion via s_wait_tensorcnt.
// ---------------------------------------------------------------------------
__device__ inline void tdm_load_2d_bf16(unsigned int lds_byte_addr,
                                        const void* gaddr,
                                        unsigned int tile_d0,      // elems/row
                                        unsigned int tile_d1,      // rows
                                        unsigned int tensor_d0,
                                        unsigned int tensor_d1,
                                        unsigned long long stride0_elems) {
    unsigned long long ga = (unsigned long long)(uintptr_t)gaddr;
    u32x4 g0;
    g0.x = 0x1u;                                        // count=1 (user D#)
    g0.y = lds_byte_addr;                               // lds_addr [63:32]
    g0.z = (unsigned int)(ga & 0xFFFFFFFFu);            // global_addr[31:0]
    g0.w = (unsigned int)((ga >> 32) & 0x01FFFFFFu)     // global_addr[56:32]
         | 0x80000000u;                                 // type=2 ("image")
    i32x8 g1;
    g1.s0 = 0x00010000;                                 // wg_mask=0, data_size=1(2B)
    g1.s1 = (int)((tensor_d0 & 0xFFFFu) << 16);         // tensor_dim0[15:0]
    g1.s2 = (int)(((tensor_d0 >> 16) & 0xFFFFu) |
                  ((tensor_d1 & 0xFFFFu) << 16));       // dim0[31:16] | dim1[15:0]
    g1.s3 = (int)(((tensor_d1 >> 16) & 0xFFFFu) |
                  ((tile_d0 & 0xFFFFu) << 16));         // dim1[31:16] | tile_dim0
    g1.s4 = (int)(tile_d1 & 0xFFFFu);                   // tile_dim1, tile_dim2=0
    g1.s5 = (int)(unsigned int)(stride0_elems & 0xFFFFFFFFull);    // stride0[31:0]
    g1.s6 = (int)(unsigned int)((stride0_elems >> 32) & 0xFFFFull);// stride0[47:32]
    g1.s7 = 0;
    i32x4 z4 = {0, 0, 0, 0};
    i32x8 z8 = {0, 0, 0, 0, 0, 0, 0, 0};
    __builtin_amdgcn_tensor_load_to_lds(g0, g1, z4, z4, z8, 0);
}

__device__ inline void tdm_wait() {
    __builtin_amdgcn_s_wait_tensorcnt(0);
}

// ---------------------------------------------------------------------------
// Kernel 1: QKV projection. x[8192,1024] f32 @ w_qkv[1024,3072] f32 + b_qkv.
// Output scattered into q/k/v bf16 slabs laid out [B,H,T,D].
// Block: 256 threads (8 waves), tile 128(M) x 128(N), K-step 32.
// ---------------------------------------------------------------------------
__global__ __launch_bounds__(256)
void qkv_gemm_kernel(const float* __restrict__ X,
                     const float* __restrict__ W,
                     const float* __restrict__ bias,
                     unsigned short* __restrict__ qkv_ws) {
    __shared__ __align__(16) unsigned short ldsA[128 * 32];   // [m][k] bf16
    __shared__ __align__(16) unsigned short ldsB[128 * 32];   // [n][k] bf16 (transposed)

    const int tid  = threadIdx.x;
    const int lane = tid & 31;
    const int wave = tid >> 5;
    const int hf   = (lane >> 4) & 1;
    const int lm   = lane & 15;
    const int wm   = wave >> 1;      // 0..3
    const int wn   = wave & 1;       // 0..1
    const int m0   = blockIdx.y * 128;
    const int n0   = blockIdx.x * 128;

    v8f acc[2][4];
#pragma unroll
    for (int mi = 0; mi < 2; ++mi)
#pragma unroll
        for (int ni = 0; ni < 4; ++ni) acc[mi][ni] = v8f_zero();

    for (int kk0 = 0; kk0 < 1024; kk0 += 32) {
        if (kk0 + 32 < 1024) {
            // Cover latency of the next tile (global_prefetch_b8).
            __builtin_prefetch(&X[(size_t)(m0 + (tid >> 3)) * 1024 + kk0 + 32 + (tid & 7) * 4], 0, 0);
            __builtin_prefetch(&W[(size_t)(kk0 + 32 + (tid >> 5)) * 3072 + n0 + (tid & 31) * 4], 0, 0);
        }
        // Stage A tile 128x32 f32 -> bf16, row-major K-contiguous.
#pragma unroll
        for (int i = 0; i < 4; ++i) {
            int idx = tid + 256 * i;           // 0..1023 float4's
            int row = idx >> 3;                // 0..127
            int c4  = idx & 7;                 // 0..7
            const float4 f = *(const float4*)&X[(size_t)(m0 + row) * 1024 + kk0 + c4 * 4];
            uint2 u;
            u.x = pack2_bf16(f.x, f.y);
            u.y = pack2_bf16(f.z, f.w);
            *(uint2*)&ldsA[row * 32 + c4 * 4] = u;
        }
        // Stage B tile 32x128 f32 -> bf16 transposed to [n][k].
#pragma unroll
        for (int i = 0; i < 4; ++i) {
            int idx = tid + 256 * i;           // 0..1023 float4's
            int row = idx >> 5;                // k: 0..31
            int c4  = idx & 31;                // n-quad: 0..31
            const float4 g = *(const float4*)&W[(size_t)(kk0 + row) * 3072 + n0 + c4 * 4];
            ldsB[(c4 * 4 + 0) * 32 + row] = f32_to_bf16(g.x);
            ldsB[(c4 * 4 + 1) * 32 + row] = f32_to_bf16(g.y);
            ldsB[(c4 * 4 + 2) * 32 + row] = f32_to_bf16(g.z);
            ldsB[(c4 * 4 + 3) * 32 + row] = f32_to_bf16(g.w);
        }
        __syncthreads();

        FragBF af[2], bfg[4];
#pragma unroll
        for (int mi = 0; mi < 2; ++mi) {
            int r = 32 * wm + 16 * mi + lm;
            af[mi].q[0] = *(const uint4*)&ldsA[r * 32 + 8 * hf];
            af[mi].q[1] = *(const uint4*)&ldsA[r * 32 + 8 * hf + 16];
        }
#pragma unroll
        for (int ni = 0; ni < 4; ++ni) {
            int r = 64 * wn + 16 * ni + lm;
            bfg[ni].q[0] = *(const uint4*)&ldsB[r * 32 + 8 * hf];
            bfg[ni].q[1] = *(const uint4*)&ldsB[r * 32 + 8 * hf + 16];
        }
#pragma unroll
        for (int mi = 0; mi < 2; ++mi)
#pragma unroll
            for (int ni = 0; ni < 4; ++ni)
                acc[mi][ni] = wmma_bf16(af[mi], bfg[ni], acc[mi][ni]);
        __syncthreads();
    }

    // Epilogue: bias + scatter into q/k/v [B,H,T,D] bf16 slabs.
    const size_t SLAB = (size_t)4 * 16 * 2048 * 64;    // 8,388,608 elements
#pragma unroll
    for (int mi = 0; mi < 2; ++mi)
#pragma unroll
        for (int ni = 0; ni < 4; ++ni)
#pragma unroll
            for (int i = 0; i < 8; ++i) {
                int mg = m0 + 32 * wm + 16 * mi + i + 8 * hf;   // 0..8191
                int ng = n0 + 64 * wn + 16 * ni + lm;           // 0..3071
                float val = acc[mi][ni][i] + bias[ng];
                int which = ng >> 10;        // 0=q 1=k 2=v
                int cc    = ng & 1023;
                int h     = cc >> 6;
                int d     = cc & 63;
                int b     = mg >> 11;
                int t     = mg & 2047;
                qkv_ws[which * SLAB + ((size_t)(b * 16 + h) * 2048 + t) * 64 + d] =
                    f32_to_bf16(val);
            }
}

// ---------------------------------------------------------------------------
// Kernel 2: flash attention with online softmax.
// Grid: (T/64, B*H). Block: 128 threads = 4 waves; wave owns 16 query rows.
// K tile is DMA'd into LDS by the Tensor Data Mover (wave 0 issues the D#),
// overlapping with the VALU/LDS work of transposing the V tile.
// ---------------------------------------------------------------------------
__global__ __launch_bounds__(128)
void attn_kernel(const unsigned short* __restrict__ Q,
                 const unsigned short* __restrict__ K,
                 const unsigned short* __restrict__ V,
                 unsigned short* __restrict__ ATT) {
    __shared__ __align__(16) unsigned short ldsK[64 * 64];     // [key][d]
    __shared__ __align__(16) unsigned short ldsVt[64 * 64];    // [d][key]
    __shared__ __align__(16) unsigned short ldsP[4][16 * 64];  // per-wave [q][key]

    const int tid  = threadIdx.x;
    const int wave = tid >> 5;
    const int lane = tid & 31;
    const int hf   = (lane >> 4) & 1;
    const int lm   = lane & 15;
    const int bh   = blockIdx.y;             // b*16 + h
    const int q0   = blockIdx.x * 64;
    const int b    = bh >> 4;
    const int h    = bh & 15;
    const unsigned int ldsK_addr = (unsigned int)(uintptr_t)&ldsK[0];

    // Q fragments resident in registers across the whole K loop.
    FragBF qa[2];
    {
        const size_t qbase = ((size_t)bh * 2048 + (q0 + 16 * wave + lm)) * 64;
#pragma unroll
        for (int f = 0; f < 2; ++f) {
            qa[f].q[0] = *(const uint4*)&Q[qbase + 32 * f + 8 * hf];
            qa[f].q[1] = *(const uint4*)&Q[qbase + 32 * f + 8 * hf + 16];
        }
    }

    v8f o[4];
    float mrow[8], lrow[8];
#pragma unroll
    for (int dt = 0; dt < 4; ++dt) o[dt] = v8f_zero();
#pragma unroll
    for (int i = 0; i < 8; ++i) { mrow[i] = -3.0e38f; lrow[i] = 0.f; }

    for (int kt0 = 0; kt0 < q0 + 64; kt0 += 64) {
        // Wave 0: kick the TDM for the K tile (64 rows x 64 bf16, contiguous).
        if (tid < 32) {
            tdm_load_2d_bf16(ldsK_addr,
                             K + ((size_t)bh * 2048 + kt0) * 64,
                             /*tile_d0=*/64, /*tile_d1=*/64,
                             /*tensor_d0=*/64, /*tensor_d1=*/2048,
                             /*stride0=*/64);
        }
        // All threads: stage V tile transposed -> ldsVt[d][key] (overlaps DMA).
#pragma unroll
        for (int i = 0; i < 8; ++i) {
            int idx = tid + 128 * i;          // 0..1023
            int key = idx >> 4;               // 0..63
            int dq  = idx & 15;               // quad of d
            uint2 u = *(const uint2*)(V + ((size_t)bh * 2048 + kt0 + key) * 64 + dq * 4);
            ldsVt[(dq * 4 + 0) * 64 + key] = (unsigned short)(u.x & 0xFFFF);
            ldsVt[(dq * 4 + 1) * 64 + key] = (unsigned short)(u.x >> 16);
            ldsVt[(dq * 4 + 2) * 64 + key] = (unsigned short)(u.y & 0xFFFF);
            ldsVt[(dq * 4 + 3) * 64 + key] = (unsigned short)(u.y >> 16);
        }
        if (tid < 32) tdm_wait();
        __syncthreads();

        // S = Q @ K^T : 4 n-tiles of 16 keys, D=64 -> 2 chained WMMAs each.
        v8f s[4];
#pragma unroll
        for (int nt = 0; nt < 4; ++nt) {
            v8f a = v8f_zero();
#pragma unroll
            for (int f = 0; f < 2; ++f) {
                FragBF kb;
                int off = (16 * nt + lm) * 64 + 32 * f + 8 * hf;
                kb.q[0] = *(const uint4*)&ldsK[off];
                kb.q[1] = *(const uint4*)&ldsK[off + 16];
                a = wmma_bf16(qa[f], kb, a);
            }
            s[nt] = a;
        }

        // Scale + causal mask.
        const float scale = 0.125f;   // 1/sqrt(64)
#pragma unroll
        for (int nt = 0; nt < 4; ++nt)
#pragma unroll
            for (int i = 0; i < 8; ++i) {
                int key = kt0 + 16 * nt + lm;
                int qg  = q0 + 16 * wave + i + 8 * hf;
                float val = s[nt][i] * scale;
                s[nt][i] = (key <= qg) ? val : -3.0e38f;
            }

        // Online softmax update (row stats replicated over the 16-lane half).
#pragma unroll
        for (int i = 0; i < 8; ++i) {
            float m = fmaxf(fmaxf(s[0][i], s[1][i]), fmaxf(s[2][i], s[3][i]));
#pragma unroll
            for (int off = 8; off >= 1; off >>= 1)
                m = fmaxf(m, __shfl_xor(m, off, 16));
            float nm    = fmaxf(mrow[i], m);
            float alpha = __expf(mrow[i] - nm);
            mrow[i] = nm;
            float rs = 0.f;
#pragma unroll
            for (int nt = 0; nt < 4; ++nt) {
                float p = __expf(s[nt][i] - nm);
                s[nt][i] = p;
                rs += p;
            }
#pragma unroll
            for (int off = 8; off >= 1; off >>= 1)
                rs += __shfl_xor(rs, off, 16);
            lrow[i] = lrow[i] * alpha + rs;
#pragma unroll
            for (int dt = 0; dt < 4; ++dt) o[dt][i] *= alpha;
        }

        // P (C-layout) -> LDS row-major so it can re-enter as an A fragment.
#pragma unroll
        for (int nt = 0; nt < 4; ++nt)
#pragma unroll
            for (int i = 0; i < 8; ++i)
                ldsP[wave][(i + 8 * hf) * 64 + 16 * nt + lm] = f32_to_bf16(s[nt][i]);

        // O += P @ V  (2 key chunks of 32, 4 d-tiles of 16).
#pragma unroll
        for (int f = 0; f < 2; ++f) {
            FragBF pa;
            pa.q[0] = *(const uint4*)&ldsP[wave][lm * 64 + 32 * f + 8 * hf];
            pa.q[1] = *(const uint4*)&ldsP[wave][lm * 64 + 32 * f + 8 * hf + 16];
#pragma unroll
            for (int dt = 0; dt < 4; ++dt) {
                FragBF vb;
                int off = (16 * dt + lm) * 64 + 32 * f + 8 * hf;
                vb.q[0] = *(const uint4*)&ldsVt[off];
                vb.q[1] = *(const uint4*)&ldsVt[off + 16];
                o[dt] = wmma_bf16(pa, vb, o[dt]);
            }
        }
        __syncthreads();
    }

    // Normalize and store att[B,T,C] bf16.
#pragma unroll
    for (int i = 0; i < 8; ++i) {
        float inv = 1.0f / lrow[i];
        int t = q0 + 16 * wave + i + 8 * hf;
#pragma unroll
        for (int dt = 0; dt < 4; ++dt) {
            int col = h * 64 + 16 * dt + lm;
            ATT[((size_t)b * 2048 + t) * 1024 + col] = f32_to_bf16(o[dt][i] * inv);
        }
    }
}

// ---------------------------------------------------------------------------
// Kernel 3: output projection. att[8192,1024] bf16 @ w_out[1024,1024] f32 + b_out
// -> d_out f32. A tile (already bf16, strided 2D) is staged by the TDM.
// ---------------------------------------------------------------------------
__global__ __launch_bounds__(256)
void out_gemm_kernel(const unsigned short* __restrict__ A,
                     const float* __restrict__ W,
                     const float* __restrict__ bias,
                     float* __restrict__ OUT) {
    __shared__ __align__(16) unsigned short ldsA[128 * 32];
    __shared__ __align__(16) unsigned short ldsB[128 * 32];

    const int tid  = threadIdx.x;
    const int lane = tid & 31;
    const int wave = tid >> 5;
    const int hf   = (lane >> 4) & 1;
    const int lm   = lane & 15;
    const int wm   = wave >> 1;
    const int wn   = wave & 1;
    const int m0   = blockIdx.y * 128;
    const int n0   = blockIdx.x * 128;
    const unsigned int ldsA_addr = (unsigned int)(uintptr_t)&ldsA[0];

    v8f acc[2][4];
#pragma unroll
    for (int mi = 0; mi < 2; ++mi)
#pragma unroll
        for (int ni = 0; ni < 4; ++ni) acc[mi][ni] = v8f_zero();

    for (int kk0 = 0; kk0 < 1024; kk0 += 32) {
        // Wave 0: TDM 2D load of A tile: 128 rows x 32 bf16, row stride 1024.
        if (tid < 32) {
            tdm_load_2d_bf16(ldsA_addr,
                             &A[(size_t)m0 * 1024 + kk0],
                             /*tile_d0=*/32, /*tile_d1=*/128,
                             /*tensor_d0=*/1024, /*tensor_d1=*/8192,
                             /*stride0=*/1024);
        }
        if (kk0 + 32 < 1024) {
            __builtin_prefetch(&W[(size_t)(kk0 + 32 + (tid >> 5)) * 1024 + n0 + (tid & 31) * 4], 0, 0);
        }
        // All threads: stage B tile 32x128 f32 -> bf16 transposed [n][k].
#pragma unroll
        for (int i = 0; i < 4; ++i) {
            int idx = tid + 256 * i;
            int row = idx >> 5;               // k: 0..31
            int c4  = idx & 31;               // n-quad
            const float4 g = *(const float4*)&W[(size_t)(kk0 + row) * 1024 + n0 + c4 * 4];
            ldsB[(c4 * 4 + 0) * 32 + row] = f32_to_bf16(g.x);
            ldsB[(c4 * 4 + 1) * 32 + row] = f32_to_bf16(g.y);
            ldsB[(c4 * 4 + 2) * 32 + row] = f32_to_bf16(g.z);
            ldsB[(c4 * 4 + 3) * 32 + row] = f32_to_bf16(g.w);
        }
        if (tid < 32) tdm_wait();
        __syncthreads();

        FragBF af[2], bfg[4];
#pragma unroll
        for (int mi = 0; mi < 2; ++mi) {
            int r = 32 * wm + 16 * mi + lm;
            af[mi].q[0] = *(const uint4*)&ldsA[r * 32 + 8 * hf];
            af[mi].q[1] = *(const uint4*)&ldsA[r * 32 + 8 * hf + 16];
        }
#pragma unroll
        for (int ni = 0; ni < 4; ++ni) {
            int r = 64 * wn + 16 * ni + lm;
            bfg[ni].q[0] = *(const uint4*)&ldsB[r * 32 + 8 * hf];
            bfg[ni].q[1] = *(const uint4*)&ldsB[r * 32 + 8 * hf + 16];
        }
#pragma unroll
        for (int mi = 0; mi < 2; ++mi)
#pragma unroll
            for (int ni = 0; ni < 4; ++ni)
                acc[mi][ni] = wmma_bf16(af[mi], bfg[ni], acc[mi][ni]);
        __syncthreads();
    }

#pragma unroll
    for (int mi = 0; mi < 2; ++mi)
#pragma unroll
        for (int ni = 0; ni < 4; ++ni)
#pragma unroll
            for (int i = 0; i < 8; ++i) {
                int mg = m0 + 32 * wm + 16 * mi + i + 8 * hf;
                int ng = n0 + 64 * wn + 16 * ni + lm;
                OUT[(size_t)mg * 1024 + ng] = acc[mi][ni][i] + bias[ng];
            }
}

// ---------------------------------------------------------------------------
extern "C" void kernel_launch(void* const* d_in, const int* in_sizes, int n_in,
                              void* d_out, int out_size, void* d_ws, size_t ws_size,
                              hipStream_t stream) {
    const float* x     = (const float*)d_in[0];
    const float* w_qkv = (const float*)d_in[1];
    const float* b_qkv = (const float*)d_in[2];
    const float* w_out = (const float*)d_in[3];
    const float* b_out = (const float*)d_in[4];
    float* out = (float*)d_out;

    unsigned short* ws = (unsigned short*)d_ws;
    const size_t SLAB = (size_t)4 * 16 * 2048 * 64;   // 8,388,608 bf16 elements
    unsigned short* qws = ws;                          // q [B,H,T,D]
    unsigned short* kws = ws + SLAB;                   // k
    unsigned short* vws = ws + 2 * SLAB;               // v
    unsigned short* att = ws + 3 * SLAB;               // att [B,T,C]

    // k1: qkv projection  (M=8192, N=3072, K=1024)
    qkv_gemm_kernel<<<dim3(24, 64), 256, 0, stream>>>(x, w_qkv, b_qkv, ws);
    // k2: flash attention (per 64-query tile per (b,h))
    attn_kernel<<<dim3(32, 64), 128, 0, stream>>>(qws, kws, vws, att);
    // k3: output projection (M=8192, N=1024, K=1024)
    out_gemm_kernel<<<dim3(8, 64), 256, 0, stream>>>(att, w_out, b_out, out);
}
